// LSTMPosUniversalModel_8194797601539
// MI455X (gfx1250) — matrix operand order repaired and checked by reading, not compiled, not run on hardware
//
#include <hip/hip_runtime.h>
#include <hip/hip_bf16.h>
#include <math.h>

// ---------------------------------------------------------------------------
// LSTM POS-routed model for MI455X (gfx1250, wave32, WMMA + async-to-LDS).
//   B=64 T=256 E=300(pad 320) H=512 4H=2048 NTAGS=17 DFF=256 NCLS=18
// Phase 2: gates_in[t] = x_t @ W_ih[tag_t]^T + biases (parallel WMMA).
// Phase 3: persistent 16-WG kernel; per step WG k computes the 128 gate rows
//          feeding its 32 h-columns (WMMA over K=512, hard-pipelined via
//          sched_group_barrier), while its gates_in[t] slice streams into LDS
//          via GLOBAL_LOAD_ASYNC_TO_LDS_B128 (ASYNCcnt); accumulate in LDS,
//          LSTM nonlinearities, publish h slice, one device barrier per step
//          (+ global_prefetch_b8 of next tag's weights during the wait).
// ---------------------------------------------------------------------------

typedef __attribute__((ext_vector_type(16))) _Float16 v16h;
typedef __attribute__((ext_vector_type(8)))  _Float16 v8h;
typedef __attribute__((ext_vector_type(8)))  float    v8f;
typedef __attribute__((ext_vector_type(4)))  int      v4i;

#define B_    64
#define T_    256
#define E_    300
#define EP    320
#define H_    512
#define G4    2048
#define NT    17
#define DFF_  256
#define NCLS_ 18

#if defined(__has_builtin)
#if __has_builtin(__builtin_amdgcn_global_load_async_to_lds_b128) && \
    __has_builtin(__builtin_amdgcn_s_wait_asynccnt)
#define USE_ASYNC 1
#endif
#if __has_builtin(__builtin_amdgcn_sched_group_barrier)
#define USE_SCHED 1
#endif
#endif

// ---- WMMA fragment loaders (per CDNA5 ISA 7.12.2 layouts) -----------------
__device__ __forceinline__ v16h load_a_frag(const _Float16* base, int ld,
                                            int row0, int k0) {
  const int lane = threadIdx.x & 31;
  const int r  = row0 + (lane & 15);
  const int ko = (lane >> 4) * 8;
  const _Float16* p = base + (size_t)r * ld + k0 + ko;
  union { v16h v; v8h h[2]; } u;
  u.h[0] = *(const v8h*)p;
  u.h[1] = *(const v8h*)(p + 16);
  return u.v;
}

__device__ __forceinline__ v16h load_b_frag(const _Float16* base, int ld,
                                            int col0, int k0) {
  const int lane = threadIdx.x & 31;
  const int r  = col0 + (lane & 15);
  const int ko = (lane >> 4) * 8;
  const _Float16* p = base + (size_t)r * ld + k0 + ko;
  union { v16h v; v8h h[2]; } u;
  u.h[0] = *(const v8h*)p;
  u.h[1] = *(const v8h*)(p + 16);
  return u.v;
}

__device__ __forceinline__ v8f wmma16(v16h a, v16h b, v8f c) {
  return __builtin_amdgcn_wmma_f32_16x16x32_f16(false, a, false, b,
                                                (short)0, c, false, false);
}

// Fully-unrolled tile GEMM with a HARD depth-2 software pipeline imposed via
// sched_group_barrier: the scheduler must place iteration k+1's VMEM reads
// before iteration k's WMMAs (masks: 0x20 = VMEM read, 0x8 = MFMA/WMMA).
template <int KTOT, int NB>
__device__ __forceinline__ void gemm_tiles(const _Float16* __restrict__ Abase, int lda, int arow,
                                           const _Float16* __restrict__ Bbase, int ldb,
                                           const int* cols, v8f* acc) {
  v16h a[2];
  v16h b[2][NB];
  a[0] = load_a_frag(Abase, lda, arow, 0);
#pragma unroll
  for (int i = 0; i < NB; ++i) b[0][i] = load_b_frag(Bbase, ldb, cols[i], 0);
#pragma unroll
  for (int kk = 0; kk < KTOT; kk += 32) {
    const int s  = (kk >> 5) & 1;
    const int sn = s ^ 1;
    if (kk + 32 < KTOT) {
      a[sn] = load_a_frag(Abase, lda, arow, kk + 32);
#pragma unroll
      for (int i = 0; i < NB; ++i)
        b[sn][i] = load_b_frag(Bbase, ldb, cols[i], kk + 32);
    }
#pragma unroll
    for (int i = 0; i < NB; ++i) acc[i] = wmma16(a[s], b[s][i], acc[i]);
  }
#ifdef USE_SCHED
  constexpr int NI = KTOT / 32;     // k-iterations
  constexpr int LD = 2 + 2 * NB;    // b128 loads per iteration (A: 2, B: 2 each)
  __builtin_amdgcn_sched_group_barrier(0x020, LD, 0);   // iter 0 loads
#pragma unroll
  for (int it = 0; it < NI - 1; ++it) {
    __builtin_amdgcn_sched_group_barrier(0x020, LD, 0); // iter it+1 loads
    __builtin_amdgcn_sched_group_barrier(0x008, NB, 0); // iter it  WMMAs
  }
  __builtin_amdgcn_sched_group_barrier(0x008, NB, 0);   // drain last WMMAs
#endif
}

__device__ __forceinline__ float sigm(float x) { return 1.0f / (1.0f + __expf(-x)); }

// ---- device-scope grid barrier (16 co-resident persistent blocks) ---------
__device__ __forceinline__ void grid_barrier(unsigned* cnt, unsigned nb) {
  __threadfence();
  __syncthreads();
  if (threadIdx.x == 0) {
    unsigned ticket = __hip_atomic_fetch_add(cnt, 1u, __ATOMIC_ACQ_REL,
                                             __HIP_MEMORY_SCOPE_AGENT);
    unsigned target = (ticket / nb + 1u) * nb;
    while (__hip_atomic_load(cnt, __ATOMIC_ACQUIRE,
                             __HIP_MEMORY_SCOPE_AGENT) < target) {
      __builtin_amdgcn_s_sleep(2);
    }
  }
  __syncthreads();
  __threadfence();
}

// ---- Phase 1: prep kernels ------------------------------------------------
__global__ void k_init(_Float16* hbuf, unsigned* cnt) {
  int i = blockIdx.x * blockDim.x + threadIdx.x;
  if (i == 0) *cnt = 0u;
  if (i < B_ * H_) hbuf[i] = (_Float16)0.0f;
}

__global__ void k_cvt_pad(const float* __restrict__ src, _Float16* __restrict__ dst,
                          long rows) {
  long total = rows * EP;
  for (long i = (long)blockIdx.x * blockDim.x + threadIdx.x; i < total;
       i += (long)gridDim.x * blockDim.x) {
    long row = i / EP; int e = (int)(i % EP);
    dst[i] = (e < E_) ? (_Float16)src[row * E_ + e] : (_Float16)0.0f;
  }
}

__global__ void k_cvt(const float* __restrict__ src, _Float16* __restrict__ dst, long n) {
  for (long i = (long)blockIdx.x * blockDim.x + threadIdx.x; i < n;
       i += (long)gridDim.x * blockDim.x)
    dst[i] = (_Float16)src[i];
}

__global__ void k_gather_x(const int* __restrict__ tokens, const float* __restrict__ emb,
                           _Float16* __restrict__ xh) {
  long total = (long)T_ * B_ * EP;
  for (long i = (long)blockIdx.x * blockDim.x + threadIdx.x; i < total;
       i += (long)gridDim.x * blockDim.x) {
    int e = (int)(i % EP);
    long tb = i / EP;
    int b = (int)(tb % B_);
    int t = (int)(tb / B_);
    int tok = tokens[b * T_ + t];
    xh[i] = (e < E_) ? (_Float16)emb[(size_t)tok * E_ + e] : (_Float16)0.0f;
  }
}

// ---- Phase 2: input projection --------------------------------------------
__global__ __launch_bounds__(256)
__attribute__((amdgpu_waves_per_eu(2)))
void k_gemm_input(
    const int* __restrict__ tags, const _Float16* __restrict__ xh,
    const _Float16* __restrict__ Wih, const float* __restrict__ bih,
    const float* __restrict__ bhh, float* __restrict__ gin) {
  const int t     = blockIdx.y;
  const int nbase = blockIdx.x * 128;
  const int tid   = threadIdx.x;
  const int lane  = tid & 31;
  const int wv    = tid >> 5;
  const int mblk  = wv & 3;
  const int nblk0 = (wv >> 2) * 4;
  const int tag   = tags[t];
  const _Float16* A  = xh + (size_t)t * B_ * EP;
  const _Float16* Bm = Wih + (size_t)tag * G4 * EP;

  int col[4];
#pragma unroll
  for (int i = 0; i < 4; ++i) col[i] = nbase + 16 * (nblk0 + i);

  v8f acc[4] = {};
  gemm_tiles<EP, 4>(A, EP, 16 * mblk, Bm, EP, col, acc);

  float* gt = gin + (size_t)t * B_ * G4;
#pragma unroll
  for (int i = 0; i < 4; ++i) {
    int n = col[i] + (lane & 15);
    float bias = bih[tag * G4 + n] + bhh[tag * G4 + n];
    union { v8f v; float f[8]; } ua; ua.v = acc[i];
#pragma unroll
    for (int r = 0; r < 8; ++r) {
      int b = 16 * mblk + r + 8 * (lane >> 4);
      gt[(size_t)b * G4 + n] = ua.f[r] + bias;
    }
  }
}

// ---- Phase 3: persistent recurrent kernel ---------------------------------
__global__ __launch_bounds__(256)
__attribute__((amdgpu_waves_per_eu(2)))
void k_lstm_seq(
    const int* __restrict__ tags, const _Float16* __restrict__ Whh,
    const float* __restrict__ gin, _Float16* __restrict__ hbuf,
    unsigned* __restrict__ bar) {
  __shared__ float gbuf[64 * 128];   // 32 KB: async-staged gates_in, then gates
  __shared__ float c_lds[64 * 32];   //  8 KB: cell state
  const int tid   = threadIdx.x;
  const int lane  = tid & 31;
  const int wv    = tid >> 5;
  const int mblk  = wv & 3;
  const int nblk0 = (wv >> 2) * 4;
  const int wg    = blockIdx.x;      // 0..15

  for (int e = tid; e < 64 * 32; e += 256) c_lds[e] = 0.0f;
  __syncthreads();

  int wrow[4];
#pragma unroll
  for (int i = 0; i < 4; ++i) {
    int nblk = nblk0 + i;
    wrow[i] = ((nblk >> 1) * H_) + 32 * wg + 16 * (nblk & 1);
  }

  for (int t = 0; t < T_; ++t) {
    const int tag = tags[t];
    const _Float16* Wt = Whh + (size_t)tag * G4 * H_;

    // kick off async staging of this step's gates_in slice (32 KB) into LDS;
    // the ASYNC engine fills gbuf while the recurrent GEMM runs below.
    {
      const float* gt = gin + (size_t)t * B_ * G4;
#pragma unroll
      for (int q = 0; q < 8; ++q) {
        int idx  = tid + q * 256;            // 0..2047 16B transfers
        int b    = idx >> 5;
        int rem  = idx & 31;
        int g    = rem >> 3;
        int part = rem & 7;
        const float* gp = gt + (size_t)b * G4 + g * H_ + 32 * wg + part * 4;
        float* lp = &gbuf[b * 128 + g * 32 + part * 4];
#ifdef USE_ASYNC
        __builtin_amdgcn_global_load_async_to_lds_b128(
            (__attribute__((address_space(1))) v4i*)gp,
            (__attribute__((address_space(3))) v4i*)lp, 0, 0);
#else
        *(float4*)lp = *(const float4*)gp;
#endif
      }
    }
#ifdef USE_SCHED
    __builtin_amdgcn_sched_barrier(0);   // keep async issue ahead of the GEMM region
#endif

    // recurrent GEMM over K=512 (hard-pipelined fragment loads)
    v8f acc[4] = {};
    gemm_tiles<H_, 4>(hbuf, H_, 16 * mblk, Wt, H_, wrow, acc);

#ifdef USE_ASYNC
    __builtin_amdgcn_s_wait_asynccnt(0);
#endif
    __syncthreads();   // gbuf fully staged by all waves

    // accumulate WMMA results into the staged gate pre-activations (in LDS)
#pragma unroll
    for (int i = 0; i < 4; ++i) {
      int nblk = nblk0 + i;
      int g    = nblk >> 1;
      int jloc = 16 * (nblk & 1) + (lane & 15);
      int nloc = g * 32 + jloc;
      union { v8f v; float f[8]; } ua; ua.v = acc[i];
#pragma unroll
      for (int r = 0; r < 8; ++r) {
        int b = 16 * mblk + r + 8 * (lane >> 4);
        gbuf[b * 128 + nloc] += ua.f[r];
      }
    }
    __syncthreads();

    for (int e = tid; e < 64 * 32; e += 256) {
      int b = e >> 5, j = e & 31;
      float iv = sigm(gbuf[b * 128 + j]);
      float fv = sigm(gbuf[b * 128 + 32 + j]);
      float gv = tanhf(gbuf[b * 128 + 64 + j]);
      float ov = sigm(gbuf[b * 128 + 96 + j]);
      float c  = fv * c_lds[e] + iv * gv;
      c_lds[e] = c;
      hbuf[b * H_ + 32 * wg + j] = (_Float16)(ov * tanhf(c));
    }

    // warm caches for next step's W_hh rows while the barrier settles
    if (t + 1 < T_) {
      const int tagn = tags[t + 1];
      const _Float16* Wn = Whh + (size_t)tagn * G4 * H_;
      for (int rr = wv; rr < 128; rr += 8) {
        int g = rr >> 5, j = rr & 31;
        const _Float16* p = Wn + (size_t)(g * H_ + 32 * wg + j) * H_ + lane * 16;
        __builtin_prefetch(p, 0, 1);   // -> global_prefetch_b8
      }
    }

    grid_barrier(bar, gridDim.x);
  }
}

// ---- Phase 4: dense head --------------------------------------------------
__global__ __launch_bounds__(256)
__attribute__((amdgpu_waves_per_eu(2)))
void k_head1(
    const _Float16* __restrict__ hbuf, const _Float16* __restrict__ W1h,
    const float* __restrict__ b1, float* __restrict__ hid) {
  const int tid = threadIdx.x, lane = tid & 31, wv = tid >> 5;
  const int mblk  = wv & 3;
  const int nblk0 = (wv >> 2) * 8;
  int col[8];
#pragma unroll
  for (int i = 0; i < 8; ++i) col[i] = 16 * (nblk0 + i);

  v8f acc[8] = {};
  gemm_tiles<H_, 8>(hbuf, H_, 16 * mblk, W1h, H_, col, acc);

#pragma unroll
  for (int i = 0; i < 8; ++i) {
    int n = col[i] + (lane & 15);
    float bias = b1[n];
    union { v8f v; float f[8]; } ua; ua.v = acc[i];
#pragma unroll
    for (int r = 0; r < 8; ++r) {
      int b = 16 * mblk + r + 8 * (lane >> 4);
      float v = ua.f[r] + bias;
      hid[b * DFF_ + n] = v > 0.0f ? v : 0.0f;
    }
  }
}

__global__ void k_head2(const float* __restrict__ hid, const float* __restrict__ W2,
                        const float* __restrict__ b2, float* __restrict__ out) {
  int idx = blockIdx.x * blockDim.x + threadIdx.x;
  if (idx >= B_ * NCLS_) return;
  int b = idx / NCLS_, cls = idx % NCLS_;
  float acc = b2[cls];
  const float* hp = hid + b * DFF_;
  const float* wp = W2 + cls * DFF_;
  for (int d = 0; d < DFF_; ++d) acc += hp[d] * wp[d];
  out[idx] = acc;
}

// ---------------------------------------------------------------------------
extern "C" void kernel_launch(void* const* d_in, const int* in_sizes, int n_in,
                              void* d_out, int out_size, void* d_ws, size_t ws_size,
                              hipStream_t stream) {
  const int*   tokens = (const int*)d_in[0];
  const int*   tags   = (const int*)d_in[1];
  const float* emb    = (const float*)d_in[2];
  const float* W_ih   = (const float*)d_in[3];
  const float* W_hh   = (const float*)d_in[4];
  const float* b_ih   = (const float*)d_in[5];
  const float* b_hh   = (const float*)d_in[6];
  const float* W1     = (const float*)d_in[7];
  const float* b1     = (const float*)d_in[8];
  const float* W2     = (const float*)d_in[9];
  const float* b2     = (const float*)d_in[10];
  float* out = (float*)d_out;

  size_t o = 0;
  auto take = [&](size_t bytes) { size_t r = o; o = (o + bytes + 255) & ~(size_t)255; return r; };
  char* ws = (char*)d_ws;
  _Float16* Wih_h = (_Float16*)(ws + take((size_t)NT * G4 * EP * 2));
  _Float16* Whh_h = (_Float16*)(ws + take((size_t)NT * G4 * H_ * 2));
  _Float16* xh    = (_Float16*)(ws + take((size_t)T_ * B_ * EP * 2));
  float*    gin   = (float*)   (ws + take((size_t)T_ * B_ * G4 * 4));
  _Float16* hbuf  = (_Float16*)(ws + take((size_t)B_ * H_ * 2));
  _Float16* W1h   = (_Float16*)(ws + take((size_t)DFF_ * H_ * 2));
  float*    hid   = (float*)   (ws + take((size_t)B_ * DFF_ * 4));
  unsigned* bar   = (unsigned*)(ws + take(64));

  k_init<<<(B_ * H_ + 255) / 256, 256, 0, stream>>>(hbuf, bar);
  k_cvt_pad<<<4096, 256, 0, stream>>>(W_ih, Wih_h, (long)NT * G4);
  k_cvt<<<4096, 256, 0, stream>>>(W_hh, Whh_h, (long)NT * G4 * H_);
  k_cvt<<<512, 256, 0, stream>>>(W1, W1h, (long)DFF_ * H_);
  k_gather_x<<<2048, 256, 0, stream>>>(tokens, emb, xh);

  k_gemm_input<<<dim3(16, T_), 256, 0, stream>>>(tags, xh, Wih_h, b_ih, b_hh, gin);
  k_lstm_seq<<<16, 256, 0, stream>>>(tags, Whh_h, gin, hbuf, bar);
  k_head1<<<1, 256, 0, stream>>>(hbuf, W1h, b1, hid);
  k_head2<<<(B_ * NCLS_ + 255) / 256, 256, 0, stream>>>(hid, W2, b2, out);
}
